// CompressiveDecoder_40913858461702
// MI455X (gfx1250) — matrix operand order repaired and checked by reading, not compiled, not run on hardware
//
#include <hip/hip_runtime.h>
#include <hip/hip_bf16.h>

#define NL 4
#define Bb 4
#define Ss 512
#define Dd 512
#define Hh 8
#define DHd 64
#define MEMN 512
#define CMEMN 128
#define KVN 1152
#define LLATN 256
#define FFD 2048
#define PADC (CMEMN + MEMN) /* 640 */

typedef __attribute__((ext_vector_type(16))) _Float16 v16h;
typedef __attribute__((ext_vector_type(8)))  float    v8f;
typedef long long ll;

// ---------------------------------------------------------------------------
// Batched/strided GEMM: C[b,h] = alpha * A[b,h] @ B[b,h] (+bias)
// One wave computes a 16(M)x64(N) strip: one A fragment feeds 4 WMMAs.
// TRANSB==true  : B stored N-major (B[n][k])  -> 4x b128 loads per tile
// TRANSB==false : B stored K-major (B[k][n])  -> 16x coalesced b32 loads
// A lane layout (contiguous k, matching B): one contiguous 16-float run ->
//   4x b128 loads at k0+16*half.
// ---------------------------------------------------------------------------
template <bool TRANSB, bool HASBIAS>
__global__ __launch_bounds__(64) void wmma_gemm_t(
    const float* __restrict__ A, const float* __restrict__ Bm,
    float* __restrict__ C, const float* __restrict__ bias,
    int K, ll lda, ll ldb, ll ldc,
    ll aSb, ll aSh, ll bSb, ll bSh, ll cSb, ll cSh,
    int Hdim, float alpha)
{
    int bz = blockIdx.z;
    int b  = bz / Hdim, h = bz % Hdim;
    A  += (ll)b * aSb + (ll)h * aSh;
    Bm += (ll)b * bSb + (ll)h * bSh;
    C  += (ll)b * cSb + (ll)h * cSh;

    int lane = threadIdx.x & 31;
    int warp = threadIdx.x >> 5;
    int half = lane >> 4;
    int r    = lane & 15;
    int tileM = blockIdx.y * 32 + warp * 16;
    int tileN = blockIdx.x * 64;

    v8f acc0 = {}, acc1 = {}, acc2 = {}, acc3 = {};
    const float* Arow = A + (ll)(tileM + r) * lda;

    for (int k0 = 0; k0 < K; k0 += 32) {
        const float* ap = Arow + k0 + half * 16;
        float4 a0 = *(const float4*)(ap);
        float4 a1 = *(const float4*)(ap + 4);
        float4 a2 = *(const float4*)(ap + 8);
        float4 a3 = *(const float4*)(ap + 12);
        v16h av;
        av[0]  = (_Float16)a0.x; av[1]  = (_Float16)a0.y;
        av[2]  = (_Float16)a0.z; av[3]  = (_Float16)a0.w;
        av[4]  = (_Float16)a1.x; av[5]  = (_Float16)a1.y;
        av[6]  = (_Float16)a1.z; av[7]  = (_Float16)a1.w;
        av[8]  = (_Float16)a2.x; av[9]  = (_Float16)a2.y;
        av[10] = (_Float16)a2.z; av[11] = (_Float16)a2.w;
        av[12] = (_Float16)a3.x; av[13] = (_Float16)a3.y;
        av[14] = (_Float16)a3.z; av[15] = (_Float16)a3.w;

#pragma unroll
        for (int j = 0; j < 4; ++j) {
            int n = tileN + j * 16 + r;
            v16h bv;
            if (TRANSB) {
                const float* bp = Bm + (ll)n * ldb + k0 + half * 16;
                float4 b0 = *(const float4*)(bp);
                float4 b1 = *(const float4*)(bp + 4);
                float4 b2 = *(const float4*)(bp + 8);
                float4 b3 = *(const float4*)(bp + 12);
                bv[0]  = (_Float16)b0.x; bv[1]  = (_Float16)b0.y;
                bv[2]  = (_Float16)b0.z; bv[3]  = (_Float16)b0.w;
                bv[4]  = (_Float16)b1.x; bv[5]  = (_Float16)b1.y;
                bv[6]  = (_Float16)b1.z; bv[7]  = (_Float16)b1.w;
                bv[8]  = (_Float16)b2.x; bv[9]  = (_Float16)b2.y;
                bv[10] = (_Float16)b2.z; bv[11] = (_Float16)b2.w;
                bv[12] = (_Float16)b3.x; bv[13] = (_Float16)b3.y;
                bv[14] = (_Float16)b3.z; bv[15] = (_Float16)b3.w;
            } else {
                const float* bp = Bm + (ll)(k0 + half * 16) * ldb + n;
#pragma unroll
                for (int t = 0; t < 16; ++t)
                    bv[t] = (_Float16)bp[(ll)t * ldb];
            }
            v8f& acc = (j == 0) ? acc0 : (j == 1) ? acc1 : (j == 2) ? acc2 : acc3;
            acc = __builtin_amdgcn_wmma_f32_16x16x32_f16(false, av, false, bv,
                                                         (short)0, acc, false, false);
        }
    }

#pragma unroll
    for (int j = 0; j < 4; ++j) {
        const v8f& acc = (j == 0) ? acc0 : (j == 1) ? acc1 : (j == 2) ? acc2 : acc3;
        int n = tileN + j * 16 + r;
        float badd = HASBIAS ? bias[n] : 0.f;
#pragma unroll
        for (int i = 0; i < 8; ++i) {
            int m = tileM + i + half * 8;
            C[(ll)m * ldc + n] = acc[i] * alpha + badd;
        }
    }
}

// out[n*K + k] = in[k*N + n]   (pre-transpose weights -> N-major for TRANSB path)
__global__ void transpose_kn(float* __restrict__ o, const float* __restrict__ in,
                             int K, int N)
{
    ll i = (ll)blockIdx.x * 256 + threadIdx.x;
    if (i >= (ll)K * N) return;
    int n = (int)(i % N);
    ll  k = i / N;
    o[(ll)n * K + k] = in[i];
}

// ---------------------------------------------------------------------------
// Softmax over one attention row (b,h,m). mode: 0=plain, 1=self (rel-pos shift
// + trg mask), 2=cross (src mask). dots layout (B,H,S,L).
// ---------------------------------------------------------------------------
__global__ void attn_softmax(float* __restrict__ dots, const float* __restrict__ pos,
                             const unsigned char* __restrict__ tmask,
                             const unsigned char* __restrict__ smask,
                             int L, int mode)
{
    int row = blockIdx.x;
    int m = row % Ss;
    int b = row / (Hh * Ss);
    float* d = dots + (ll)row * L;
    const float* p = (mode == 1) ? pos + (ll)row * L : nullptr;
    int tid = threadIdx.x;
    __shared__ float sh[256];

    float mx = -3.4e38f;
    for (int k = tid; k < L; k += 256) {
        float v = d[k];
        if (mode == 1) {
            int idx = k + Ss - 1 - m;          // shift(): out[m,k]=pos[m,k+S-1-m]
            if (idx >= 0 && idx < L) v += p[idx];
            bool ok = (k < PADC) ? true
                    : (tmask[b * Ss + m] && tmask[b * Ss + (k - PADC)]);
            if (!ok) v = -1e9f;
        } else if (mode == 2) {
            if (!smask[((ll)b * Ss + m) * LLATN + k]) v = -1e9f;
        }
        d[k] = v;
        mx = fmaxf(mx, v);
    }
    sh[tid] = mx; __syncthreads();
    for (int o = 128; o > 0; o >>= 1) { if (tid < o) sh[tid] = fmaxf(sh[tid], sh[tid + o]); __syncthreads(); }
    mx = sh[0]; __syncthreads();

    float sum = 0.f;
    for (int k = tid; k < L; k += 256) { float e = expf(d[k] - mx); d[k] = e; sum += e; }
    sh[tid] = sum; __syncthreads();
    for (int o = 128; o > 0; o >>= 1) { if (tid < o) sh[tid] += sh[tid + o]; __syncthreads(); }
    float inv = 1.f / sh[0];
    for (int k = tid; k < L; k += 256) d[k] *= inv;
}

// out = LN(in1 + in2) * g + b   (in2 nullable; out may alias in2)
__global__ void add_layernorm(float* __restrict__ out, const float* __restrict__ in1,
                              const float* __restrict__ in2,
                              const float* __restrict__ g, const float* __restrict__ bt)
{
    int row = blockIdx.x, tid = threadIdx.x;
    const float* a = in1 + (ll)row * Dd;
    const float* c = in2 ? in2 + (ll)row * Dd : nullptr;
    __shared__ float sh[256];
    float s = 0.f;
    for (int k = tid; k < Dd; k += 256) s += a[k] + (c ? c[k] : 0.f);
    sh[tid] = s; __syncthreads();
    for (int o = 128; o > 0; o >>= 1) { if (tid < o) sh[tid] += sh[tid + o]; __syncthreads(); }
    float mu = sh[0] / Dd; __syncthreads();
    float s2 = 0.f;
    for (int k = tid; k < Dd; k += 256) { float v = a[k] + (c ? c[k] : 0.f) - mu; s2 += v * v; }
    sh[tid] = s2; __syncthreads();
    for (int o = 128; o > 0; o >>= 1) { if (tid < o) sh[tid] += sh[tid + o]; __syncthreads(); }
    float rs = rsqrtf(sh[0] / Dd + 1e-5f);
    float* o2 = out + (ll)row * Dd;
    for (int k = tid; k < Dd; k += 256) {
        float v = a[k] + (c ? c[k] : 0.f);
        o2[k] = (v - mu) * rs * g[k] + bt[k];
    }
}

__global__ void embed_lookup(float* __restrict__ x, const float* __restrict__ emb,
                             const int* __restrict__ trg)
{
    ll i = (ll)blockIdx.x * 256 + threadIdx.x;
    if (i >= (ll)Bb * Ss * Dd) return;
    int d = (int)(i % Dd);
    ll r = i / Dd;
    x[i] = emb[(ll)trg[r] * Dd + d];
}

__global__ void concat_kv(float* __restrict__ kv, const float* __restrict__ cm,
                          const float* __restrict__ mm, const float* __restrict__ x)
{
    ll i = (ll)blockIdx.x * 256 + threadIdx.x;
    if (i >= (ll)Bb * KVN * Dd) return;
    int d = (int)(i % Dd);
    ll rest = i / Dd;
    int j = (int)(rest % KVN);
    int b = (int)(rest / KVN);
    float v;
    if (j < CMEMN)      v = cm[((ll)b * CMEMN + j) * Dd + d];
    else if (j < PADC)  v = mm[((ll)b * MEMN + (j - CMEMN)) * Dd + d];
    else                v = x[((ll)b * Ss + (j - PADC)) * Dd + d];
    kv[i] = v;
}

// wtT[o*2048 + (r*512+d)] = conv_w[(o*512+d)*4 + r]  (N-major for TRANSB path)
__global__ void repack_convw(float* __restrict__ wt, const float* __restrict__ w)
{
    ll i = (ll)blockIdx.x * 256 + threadIdx.x;
    if (i >= (ll)Dd * Dd * 4) return;
    int rd = (int)(i % (Dd * 4));
    int o  = (int)(i / (Dd * 4));
    int r  = rd / Dd;
    int d  = rd % Dd;
    wt[i] = w[((ll)o * Dd + d) * 4 + r];
}

__global__ void gelu_inplace(float* __restrict__ x, ll n)
{
    ll i = (ll)blockIdx.x * 256 + threadIdx.x;
    if (i >= n) return;
    float v = x[i];
    x[i] = 0.5f * v * (1.f + erff(v * 0.70710678118654752f));
}

__global__ void add_inplace(float* __restrict__ x, const float* __restrict__ y, ll n)
{
    ll i = (ll)blockIdx.x * 256 + threadIdx.x;
    if (i < n) x[i] += y[i];
}

__global__ void zero_scalar(float* p) { p[0] = 0.f; }

__global__ void mse_accum(const float* __restrict__ a, const float* __restrict__ b,
                          float* __restrict__ acc, ll n, float inv_n)
{
    int tid = threadIdx.x;
    ll i = (ll)blockIdx.x * 256 + tid;
    float s = 0.f;
    for (; i < n; i += (ll)gridDim.x * 256) { float d = a[i] - b[i]; s += d * d; }
    __shared__ float sh[256];
    sh[tid] = s; __syncthreads();
    for (int o = 128; o > 0; o >>= 1) { if (tid < o) sh[tid] += sh[tid + o]; __syncthreads(); }
    if (tid == 0) atomicAdd(acc, sh[0] * inv_n);
}

__global__ void finalize(float* __restrict__ out, const float* __restrict__ loss,
                         const float* __restrict__ x, ll n)
{
    ll i = (ll)blockIdx.x * 256 + threadIdx.x;
    if (i < n)       out[i] = x[i];
    else if (i == n) out[n] = loss[0] * (1.f / NL);
}

// ---------------------------------------------------------------------------
extern "C" void kernel_launch(void* const* d_in, const int* in_sizes, int n_in,
                              void* d_out, int out_size, void* d_ws, size_t ws_size,
                              hipStream_t stream)
{
    const int*           trg      = (const int*)d_in[0];
    const unsigned char* trg_mask = (const unsigned char*)d_in[1];
    const unsigned char* src_mask = (const unsigned char*)d_in[2];
    const float* latent  = (const float*)d_in[3];
    const float* mems    = (const float*)d_in[4];
    const float* cmems   = (const float*)d_in[5];
    const float* pos_emb = (const float*)d_in[6];
    const float* embed   = (const float*)d_in[7];
    const float* W_self  = (const float*)d_in[8];
    const float* ln1_g   = (const float*)d_in[9];
    const float* ln1_b   = (const float*)d_in[10];
    const float* conv_w  = (const float*)d_in[11];
    const float* conv_b  = (const float*)d_in[12];
    const float* W_src   = (const float*)d_in[13];
    const float* ln2_g   = (const float*)d_in[14];
    const float* ln2_b   = (const float*)d_in[15];
    const float* w1p     = (const float*)d_in[16];
    const float* b1p     = (const float*)d_in[17];
    const float* w2p     = (const float*)d_in[18];
    const float* b2p     = (const float*)d_in[19];
    float* out = (float*)d_out;

    float* Wp = (float*)d_ws;
    size_t off = 0;
    auto alloc = [&](size_t n) { float* p = Wp + off; off += n; return p; };
    const size_t XSZ = (size_t)Bb * Ss * Dd;  // 1048576
    const ll DD = (ll)Dd * Dd;
    float* xbuf  = alloc(XSZ);
    float* kvbuf = alloc((size_t)Bb * KVN * Dd);
    float* qbuf  = alloc(XSZ);
    float* kbuf  = alloc((size_t)Bb * KVN * Dd);
    float* vbuf  = alloc((size_t)Bb * KVN * Dd);
    float* dots  = alloc((size_t)Bb * Hh * Ss * KVN);
    float* posd  = alloc((size_t)Bb * Hh * Ss * KVN);
    float* att   = alloc(XSZ);
    float* proj  = alloc(XSZ);
    float* newcm = alloc((size_t)Bb * CMEMN * Dd);
    float* wtT   = alloc((size_t)Dd * Dd * 4);   // conv weight, N-major
    float* tgt   = alloc(XSZ);
    float* prd   = alloc(XSZ);
    float* ybuf  = alloc(XSZ);
    float* ff1   = alloc((size_t)Bb * Ss * FFD);
    float* ff2   = alloc(XSZ);
    // per-layer transposed weights (N-major), reused each layer
    float* wqT = alloc(DD); float* wkT = alloc(DD);
    float* wvT = alloc(DD); float* woT = alloc(DD);
    float* sqT = alloc(DD); float* skT = alloc(DD);
    float* svT = alloc(DD); float* soT = alloc(DD);
    float* w1T = alloc((size_t)Dd * FFD);
    float* w2T = alloc((size_t)Dd * FFD);
    float* loss = alloc(1);

    auto gemm = [&](const float* A, const float* B, float* C, const float* bias,
                    int M, int N, int K, ll lda, ll ldb, ll ldc,
                    ll aSb, ll aSh, ll bSb, ll bSh, ll cSb, ll cSh,
                    int batch, int Hdim, int transB, float alpha) {
        dim3 g(N / 64, M / 32, batch);
        if (transB) {
            if (bias) wmma_gemm_t<true, true><<<g, 64, 0, stream>>>(A, B, C, bias, K, lda, ldb, ldc, aSb, aSh, bSb, bSh, cSb, cSh, Hdim, alpha);
            else      wmma_gemm_t<true, false><<<g, 64, 0, stream>>>(A, B, C, bias, K, lda, ldb, ldc, aSb, aSh, bSb, bSh, cSb, cSh, Hdim, alpha);
        } else {
            if (bias) wmma_gemm_t<false, true><<<g, 64, 0, stream>>>(A, B, C, bias, K, lda, ldb, ldc, aSb, aSh, bSb, bSh, cSb, cSh, Hdim, alpha);
            else      wmma_gemm_t<false, false><<<g, 64, 0, stream>>>(A, B, C, bias, K, lda, ldb, ldc, aSb, aSh, bSb, bSh, cSb, cSh, Hdim, alpha);
        }
    };
    auto trsp = [&](float* o, const float* in, int K, int N) {
        transpose_kn<<<(int)(((ll)K * N + 255) / 256), 256, 0, stream>>>(o, in, K, N);
    };

    zero_scalar<<<1, 1, 0, stream>>>(loss);
    embed_lookup<<<(int)(XSZ / 256), 256, 0, stream>>>(xbuf, embed, trg);

    for (int i = 0; i < NL; ++i) {
        const float* mi = mems  + (size_t)i * Bb * MEMN  * Dd;
        const float* ci = cmems + (size_t)i * Bb * CMEMN * Dd;

        // ---- pre-transpose this layer's weights (N-major for b128 loads) ----
        trsp(wqT, W_self + (size_t)(i * 4 + 0) * DD, Dd, Dd);
        trsp(wkT, W_self + (size_t)(i * 4 + 1) * DD, Dd, Dd);
        trsp(wvT, W_self + (size_t)(i * 4 + 2) * DD, Dd, Dd);
        trsp(woT, W_self + (size_t)(i * 4 + 3) * DD, Dd, Dd);
        trsp(sqT, W_src  + (size_t)(i * 4 + 0) * DD, Dd, Dd);
        trsp(skT, W_src  + (size_t)(i * 4 + 1) * DD, Dd, Dd);
        trsp(svT, W_src  + (size_t)(i * 4 + 2) * DD, Dd, Dd);
        trsp(soT, W_src  + (size_t)(i * 4 + 3) * DD, Dd, Dd);
        trsp(w1T, w1p + (size_t)i * Dd * FFD, Dd, FFD);
        trsp(w2T, w2p + (size_t)i * FFD * Dd, FFD, Dd);
        repack_convw<<<(int)((size_t)Dd * Dd * 4 / 256), 256, 0, stream>>>(wtT, conv_w + (size_t)i * Dd * Dd * 4);

        // ---- self attention over [cmem | mem | x] ----
        concat_kv<<<(int)((size_t)Bb * KVN * Dd / 256), 256, 0, stream>>>(kvbuf, ci, mi, xbuf);
        gemm(xbuf,  wqT, qbuf, nullptr, Bb * Ss,  Dd, Dd, Dd, Dd, Dd, 0,0,0,0,0,0, 1, 1, 1, 1.f);
        gemm(kvbuf, wkT, kbuf, nullptr, Bb * KVN, Dd, Dd, Dd, Dd, Dd, 0,0,0,0,0,0, 1, 1, 1, 1.f);
        gemm(kvbuf, wvT, vbuf, nullptr, Bb * KVN, Dd, Dd, Dd, Dd, Dd, 0,0,0,0,0,0, 1, 1, 1, 1.f);
        gemm(qbuf, kbuf, dots, nullptr, Ss, KVN, DHd, Dd, Dd, KVN,
             (ll)Ss * Dd, DHd, (ll)KVN * Dd, DHd, (ll)Hh * Ss * KVN, (ll)Ss * KVN,
             Bb * Hh, Hh, 1, 0.125f);
        gemm(qbuf, pos_emb, posd, nullptr, Ss, KVN, DHd, Dd, DHd, KVN,
             (ll)Ss * Dd, DHd, 0, (ll)KVN * DHd, (ll)Hh * Ss * KVN, (ll)Ss * KVN,
             Bb * Hh, Hh, 1, 8.0f);
        attn_softmax<<<Bb * Hh * Ss, 256, 0, stream>>>(dots, posd, trg_mask, nullptr, KVN, 1);
        gemm(dots, vbuf, att, nullptr, Ss, DHd, KVN, KVN, Dd, Dd,
             (ll)Hh * Ss * KVN, (ll)Ss * KVN, (ll)KVN * Dd, DHd, (ll)Ss * Dd, DHd,
             Bb * Hh, Hh, 0, 1.f);
        gemm(att, woT, proj, nullptr, Bb * Ss, Dd, Dd, Dd, Dd, Dd, 0,0,0,0,0,0, 1, 1, 1, 1.f);
        add_layernorm<<<Bb * Ss, 256, 0, stream>>>(xbuf, proj, xbuf,
                                                   ln1_g + (size_t)i * Dd, ln1_b + (size_t)i * Dd);

        // ---- conv compress: new_cm = old_mem (B*128,2048) @ wt + b ----
        gemm(mi, wtT, newcm, conv_b + (size_t)i * Dd, Bb * CMEMN, Dd, Dd * 4,
             Dd * 4, Dd * 4, Dd, 0,0,0,0,0,0, 1, 1, 1, 1.f);

        // ---- reconstruction attention loss ----
        gemm(xbuf, wqT, qbuf, nullptr, Bb * Ss, Dd, Dd, Dd, Dd, Dd, 0,0,0,0,0,0, 1, 1, 1, 1.f);
        // target: attn over old_mem
        gemm(mi, wkT, kbuf, nullptr, Bb * MEMN, Dd, Dd, Dd, Dd, Dd, 0,0,0,0,0,0, 1, 1, 1, 1.f);
        gemm(mi, wvT, vbuf, nullptr, Bb * MEMN, Dd, Dd, Dd, Dd, Dd, 0,0,0,0,0,0, 1, 1, 1, 1.f);
        gemm(qbuf, kbuf, dots, nullptr, Ss, MEMN, DHd, Dd, Dd, MEMN,
             (ll)Ss * Dd, DHd, (ll)MEMN * Dd, DHd, (ll)Hh * Ss * MEMN, (ll)Ss * MEMN,
             Bb * Hh, Hh, 1, 0.125f);
        attn_softmax<<<Bb * Hh * Ss, 256, 0, stream>>>(dots, nullptr, nullptr, nullptr, MEMN, 0);
        gemm(dots, vbuf, tgt, nullptr, Ss, DHd, MEMN, MEMN, Dd, Dd,
             (ll)Hh * Ss * MEMN, (ll)Ss * MEMN, (ll)MEMN * Dd, DHd, (ll)Ss * Dd, DHd,
             Bb * Hh, Hh, 0, 1.f);
        // prediction: attn over new_cm
        gemm(newcm, wkT, kbuf, nullptr, Bb * CMEMN, Dd, Dd, Dd, Dd, Dd, 0,0,0,0,0,0, 1, 1, 1, 1.f);
        gemm(newcm, wvT, vbuf, nullptr, Bb * CMEMN, Dd, Dd, Dd, Dd, Dd, 0,0,0,0,0,0, 1, 1, 1, 1.f);
        gemm(qbuf, kbuf, dots, nullptr, Ss, CMEMN, DHd, Dd, Dd, CMEMN,
             (ll)Ss * Dd, DHd, (ll)CMEMN * Dd, DHd, (ll)Hh * Ss * CMEMN, (ll)Ss * CMEMN,
             Bb * Hh, Hh, 1, 0.125f);
        attn_softmax<<<Bb * Hh * Ss, 256, 0, stream>>>(dots, nullptr, nullptr, nullptr, CMEMN, 0);
        gemm(dots, vbuf, prd, nullptr, Ss, DHd, CMEMN, CMEMN, Dd, Dd,
             (ll)Hh * Ss * CMEMN, (ll)Ss * CMEMN, (ll)CMEMN * Dd, DHd, (ll)Ss * Dd, DHd,
             Bb * Hh, Hh, 0, 1.f);
        mse_accum<<<1024, 256, 0, stream>>>(prd, tgt, loss, (ll)XSZ, 1.0f / (float)XSZ);

        // ---- cross attention with latent ----
        gemm(xbuf,   sqT, qbuf, nullptr, Bb * Ss,    Dd, Dd, Dd, Dd, Dd, 0,0,0,0,0,0, 1, 1, 1, 1.f);
        gemm(latent, skT, kbuf, nullptr, Bb * LLATN, Dd, Dd, Dd, Dd, Dd, 0,0,0,0,0,0, 1, 1, 1, 1.f);
        gemm(latent, svT, vbuf, nullptr, Bb * LLATN, Dd, Dd, Dd, Dd, Dd, 0,0,0,0,0,0, 1, 1, 1, 1.f);
        gemm(qbuf, kbuf, dots, nullptr, Ss, LLATN, DHd, Dd, Dd, LLATN,
             (ll)Ss * Dd, DHd, (ll)LLATN * Dd, DHd, (ll)Hh * Ss * LLATN, (ll)Ss * LLATN,
             Bb * Hh, Hh, 1, 0.125f);
        attn_softmax<<<Bb * Hh * Ss, 256, 0, stream>>>(dots, nullptr, nullptr, src_mask, LLATN, 2);
        gemm(dots, vbuf, att, nullptr, Ss, DHd, LLATN, LLATN, Dd, Dd,
             (ll)Hh * Ss * LLATN, (ll)Ss * LLATN, (ll)LLATN * Dd, DHd, (ll)Ss * Dd, DHd,
             Bb * Hh, Hh, 0, 1.f);
        gemm(att, soT, xbuf, nullptr, Bb * Ss, Dd, Dd, Dd, Dd, Dd, 0,0,0,0,0,0, 1, 1, 1, 1.f);

        // ---- feed-forward ----
        add_layernorm<<<Bb * Ss, 256, 0, stream>>>(ybuf, xbuf, nullptr,
                                                   ln2_g + (size_t)i * Dd, ln2_b + (size_t)i * Dd);
        gemm(ybuf, w1T, ff1, b1p + (size_t)i * FFD, Bb * Ss, FFD, Dd,
             Dd, Dd, FFD, 0,0,0,0,0,0, 1, 1, 1, 1.f);
        gelu_inplace<<<(int)((size_t)Bb * Ss * FFD / 256), 256, 0, stream>>>(ff1, (ll)Bb * Ss * FFD);
        gemm(ff1, w2T, ff2, b2p + (size_t)i * Dd, Bb * Ss, Dd, FFD,
             FFD, FFD, Dd, 0,0,0,0,0,0, 1, 1, 1, 1.f);
        add_inplace<<<(int)(XSZ / 256), 256, 0, stream>>>(xbuf, ff2, (ll)XSZ);
    }

    finalize<<<(int)(XSZ / 256) + 1, 256, 0, stream>>>(out, loss, xbuf, (ll)XSZ);
}
